// EGNNDecoder_72584947302433
// MI455X (gfx1250) — matrix operand order repaired and checked by reading, not compiled, not run on hardware
//
#include <hip/hip_runtime.h>
#include <math.h>

typedef __attribute__((ext_vector_type(16))) _Float16 v16h;
typedef __attribute__((ext_vector_type(8)))  _Float16 v8h;
typedef __attribute__((ext_vector_type(8)))  float    v8f;

#define NB 4          // batch
#define LN_ 2048      // nodes per graph
#define LN_EPS 1e-5f

// ---------------------------------------------------------------------------
// WMMA fragment loaders (wave32, f16, 16x16x32).
//   A (16x32) from LDS tile, row stride `ld` halves (ld*2 % 16 == 0):
//     lane L: m=L&15, half=L>>4; j0..15 -> K=16*half+j
//   B (32x16) directly from global, stored transposed Wt[n][Kpad]:
//     lane L: n=L&15, half=L>>4; 16 contiguous K starting at k0+16*half
// Weights are padded in BOTH K (to Kpad) and N (to a multiple of 64) with
// zeros, so fragment loads are unconditional straight-line code.
// ---------------------------------------------------------------------------
__device__ __forceinline__ v16h frag_a(const _Float16* base, int ld) {
    int lane = threadIdx.x & 31;
    int m = lane & 15, half = lane >> 4;
    const _Float16* p = base + m * ld;
    v8h lo = *(const v8h*)(p + 16 * half);
    v8h hi = *(const v8h*)(p + 16 * half + 8);
    v16h r;
#pragma unroll
    for (int i = 0; i < 8; ++i) { r[i] = lo[i]; r[8 + i] = hi[i]; }
    return r;
}

__device__ __forceinline__ v16h frag_b_g(const _Float16* __restrict__ Wt, int rowstride,
                                         int n0, int k0) {
    int lane = threadIdx.x & 31;
    int n = n0 + (lane & 15), half = lane >> 4;
    const v8h* p = (const v8h*)(Wt + (long)n * rowstride + k0 + 16 * half);
    v8h lo = p[0], hi = p[1];
    v16h r;
#pragma unroll
    for (int i = 0; i < 8; ++i) { r[i] = lo[i]; r[8 + i] = hi[i]; }
    return r;
}

__device__ __forceinline__ void st4h(_Float16* dst, float4 v) {
    dst[0] = (_Float16)v.x; dst[1] = (_Float16)v.y;
    dst[2] = (_Float16)v.z; dst[3] = (_Float16)v.w;
}

__device__ __forceinline__ float siluf(float v) { return v / (1.f + __expf(-v)); }
__device__ __forceinline__ float sigmf(float v) { return 1.f / (1.f + __expf(-v)); }

// ---------------------------------------------------------------------------
// Weight transpose + f16 convert with zero padding:
//   Wt[l][n][k] = (half)W[l][k][n]  for n<N, k<K; else 0.  (Npad x Kpad tile)
// ---------------------------------------------------------------------------
__global__ void wconvT(const float* __restrict__ W, _Float16* __restrict__ Wt,
                       int K, int N, int Kpad, int Npad) {
    long idx = (long)blockIdx.x * 256 + threadIdx.x;
    long total = (long)Npad * Kpad;
    if (idx >= total) return;
    W  += (long)blockIdx.y * K * N;
    Wt += (long)blockIdx.y * total;
    int n = (int)(idx / Kpad), k = (int)(idx % Kpad);
    Wt[idx] = (_Float16)((k < K && n < N) ? W[(long)k * N + n] : 0.f);
}

// ---------------------------------------------------------------------------
// Generic tiled WMMA GEMM:  C[M,N] = act(A[M,K] @ W[K,N] + bias)
// 256 threads (8 waves). BM=128, BN=64, BK=32; wave w owns rows [16w,16w+16).
// A described as two row-major segments (lda==0 -> broadcast row).
// Double-buffered LDS staging: stage step i+1 while computing step i; one
// barrier per K-step. B fragments direct from transposed padded f16 weights.
// Requires: M % 128 == 0, K % 32 == 0, k0len % 32 == 0, Wt padded to >=64 cols.
// ---------------------------------------------------------------------------
template <int ACT>   // 0 none, 1 relu, 2 silu
__global__ __launch_bounds__(256) void wgemm(
    const float* __restrict__ a0, long lda0, int k0len, long bs0,
    const float* __restrict__ a1, long lda1, long bs1,
    const _Float16* __restrict__ Wt, const float* __restrict__ bias,
    float* __restrict__ C, int M, int K, int N, int Kpad, long cbs) {
    __shared__ _Float16 As[2][128 * 40];
    a0 += (long)blockIdx.z * bs0;
    a1 += (long)blockIdx.z * bs1;
    C  += (long)blockIdx.z * cbs;
    const int tid = threadIdx.x;
    const int wave = tid >> 5, lane = tid & 31;
    const int m0 = blockIdx.x * 128, n0 = blockIdx.y * 64;

    v8f acc[4];
#pragma unroll
    for (int t = 0; t < 4; ++t)
#pragma unroll
        for (int j = 0; j < 8; ++j) acc[t][j] = 0.f;

    const int r = tid >> 1, ks = (tid & 1) * 16;
    const float* p0 = a0 + (long)(m0 + r) * lda0 + ks;
    const float* p1 = a1 + (long)(m0 + r) * lda1 + ks;

    auto srcp = [&](int k0) -> const float* {
        return (k0 < k0len) ? p0 + k0 : p1 + (k0 - k0len);
    };
    auto stage = [&](int buf, const float* src) {
        const float4* q = (const float4*)src;
#pragma unroll
        for (int c = 0; c < 4; ++c)
            st4h(&As[buf][r * 40 + ks + c * 4], q[c]);
    };
    auto compute = [&](int buf, int k0) {
        v16h a = frag_a(As[buf] + (16 * wave) * 40, 40);
#pragma unroll
        for (int t = 0; t < 4; ++t) {
            v16h b = frag_b_g(Wt, Kpad, n0 + 16 * t, k0);
            acc[t] = __builtin_amdgcn_wmma_f32_16x16x32_f16(
                false, a, false, b, (short)0, acc[t], false, false);
        }
    };

    stage(0, srcp(0));
    __syncthreads();
    int i = 0;
#pragma unroll 1
    for (; 32 * (i + 1) < K; ++i) {
        stage((i + 1) & 1, srcp(32 * (i + 1)));   // overlap with compute
        compute(i & 1, 32 * i);
        __syncthreads();
    }
    compute(i & 1, 32 * i);

    const int half = lane >> 4, nl = lane & 15;
#pragma unroll
    for (int t = 0; t < 4; ++t)
#pragma unroll
        for (int vg = 0; vg < 8; ++vg) {
            int rr = m0 + 16 * wave + vg + 8 * half;
            int c = n0 + 16 * t + nl;
            if (c < N) {
                float v = acc[t][vg] + bias[c];
                if (ACT == 1) v = fmaxf(v, 0.f);
                if (ACT == 2) v = siluf(v);
                C[(long)rr * N + c] = v;
            }
        }
}

// ---------------------------------------------------------------------------
// Fused edge pipeline, 32 edges / block, 256 threads (8 waves: wv_m=wave&1,
// wv_n=wave>>1; each wave = 16 rows x 64 cols). Whole gathered A tile
// (32x544 f16) staged once; all three WMMA K-loops are barrier-free and
// straight-line. M2 tile aliases the A region (A dead after GEMM1).
//   m1 = silu([h[row]|h[col]|d2] @ pe1 + b)   (K=513 padded to 544)
//   m  = silu(m1 @ pe2 + b)   -> atomic agg[row] += m
//   u  = silu(m  @ px1 + b);  w = u . px2 + b -> atomic xupd[row] += w*rel
// ---------------------------------------------------------------------------
__global__ __launch_bounds__(256) void edge_fused(
    const float* __restrict__ h, const float* __restrict__ x,
    const int* __restrict__ erow, const int* __restrict__ ecol,
    const _Float16* __restrict__ pe1t, const float* __restrict__ pe1b,
    const _Float16* __restrict__ pe2t, const float* __restrict__ pe2b,
    const _Float16* __restrict__ px1t, const float* __restrict__ px1b,
    const float* __restrict__ px2W, const float* __restrict__ px2b,
    float* __restrict__ agg, float* __restrict__ xupd, int E) {
    __shared__ _Float16 Af[32 * 544];          // A tile; reused as M2 tile
    __shared__ _Float16 M1[32 * 264];
    __shared__ float  wsum[32];
    __shared__ int    sRow[32], sCol[32];
    __shared__ float  sRel[32 * 3], sD2[32];
    _Float16* M2s = Af;                        // alias (A dead after GEMM1)

    const int b = blockIdx.y;
    h    += (long)b * LN_ * 256;
    x    += (long)b * LN_ * 3;
    agg  += (long)b * LN_ * 256;
    xupd += (long)b * LN_ * 3;

    const int tid = threadIdx.x;
    const int wave = tid >> 5, lane = tid & 31;
    const int wv_m = wave & 1, wv_n = wave >> 1;
    const int half = lane >> 4, nl = lane & 15;
    const int e0 = blockIdx.x * 32;

    if (tid < 32) {
        int e = e0 + tid;
        bool ok = e < E;
        int r = ok ? erow[e] : 0;
        int c = ok ? ecol[e] : 0;
        sRow[tid] = r; sCol[tid] = c;
        float d2 = 0.f;
#pragma unroll
        for (int i = 0; i < 3; ++i) {
            float rl = ok ? (x[r * 3 + i] - x[c * 3 + i]) : 0.f;
            sRel[tid * 3 + i] = rl;
            d2 += rl * rl;
        }
        sD2[tid] = d2;
        wsum[tid] = 0.f;
    }
    __syncthreads();

    // ---- stage full A tile: 8 threads/row, 4 consecutive k each step ----
    {
        const int ar = tid >> 3, kk = (tid & 7) * 4;
        const float aD2 = sD2[ar];
        const float* hR = h + (long)sRow[ar] * 256 + kk;
        const float* hC = h + (long)sCol[ar] * 256 + kk;
        _Float16* dst = Af + ar * 544 + kk;
#pragma unroll 1
        for (int ksi = 0; ksi < 256; ksi += 32)
            st4h(dst + ksi, *(const float4*)(hR + ksi));
#pragma unroll 1
        for (int ksi = 0; ksi < 256; ksi += 32)
            st4h(dst + 256 + ksi, *(const float4*)(hC + ksi));
        st4h(dst + 512, make_float4(kk == 0 ? aD2 : 0.f, 0.f, 0.f, 0.f));
        __builtin_prefetch(pe1t + (long)(64 * wv_n) * 544, 0, 1);
    }
    __syncthreads();

    v8f acc[4];
#pragma unroll
    for (int t = 0; t < 4; ++t)
#pragma unroll
        for (int j = 0; j < 8; ++j) acc[t][j] = 0.f;

    // ---- GEMM1: K = 544, barrier-free ----
    for (int ksi = 0; ksi < 544; ksi += 32) {
        v16h a = frag_a(Af + (16 * wv_m) * 544 + ksi, 544);
#pragma unroll
        for (int t = 0; t < 4; ++t) {
            v16h bb = frag_b_g(pe1t, 544, 64 * wv_n + 16 * t, ksi);
            acc[t] = __builtin_amdgcn_wmma_f32_16x16x32_f16(
                false, a, false, bb, (short)0, acc[t], false, false);
        }
    }
    __syncthreads();   // all waves done reading Af
#pragma unroll
    for (int t = 0; t < 4; ++t)
#pragma unroll
        for (int vg = 0; vg < 8; ++vg) {
            int r = 16 * wv_m + vg + 8 * half;
            int c = 64 * wv_n + 16 * t + nl;
            M1[r * 264 + c] = (_Float16)siluf(acc[t][vg] + pe1b[c]);
            acc[t][vg] = 0.f;
        }
    __syncthreads();

    // ---- GEMM2: m = silu(m1 @ pe2), K=256, barrier-free ----
    for (int ksi = 0; ksi < 256; ksi += 32) {
        v16h a = frag_a(M1 + (16 * wv_m) * 264 + ksi, 264);
#pragma unroll
        for (int t = 0; t < 4; ++t) {
            v16h bb = frag_b_g(pe2t, 256, 64 * wv_n + 16 * t, ksi);
            acc[t] = __builtin_amdgcn_wmma_f32_16x16x32_f16(
                false, a, false, bb, (short)0, acc[t], false, false);
        }
    }
#pragma unroll
    for (int t = 0; t < 4; ++t)
#pragma unroll
        for (int vg = 0; vg < 8; ++vg) {
            int r = 16 * wv_m + vg + 8 * half;
            int c = 64 * wv_n + 16 * t + nl;
            float v = siluf(acc[t][vg] + pe2b[c]);
            M2s[r * 264 + c] = (_Float16)v;
            if (e0 + r < E) atomicAdd(&agg[(long)sRow[r] * 256 + c], v);
            acc[t][vg] = 0.f;
        }
    __syncthreads();

    // ---- GEMM3: u = silu(m @ px1), w = u . px2, barrier-free ----
    for (int ksi = 0; ksi < 256; ksi += 32) {
        v16h a = frag_a(M2s + (16 * wv_m) * 264 + ksi, 264);
#pragma unroll
        for (int t = 0; t < 4; ++t) {
            v16h bb = frag_b_g(px1t, 256, 64 * wv_n + 16 * t, ksi);
            acc[t] = __builtin_amdgcn_wmma_f32_16x16x32_f16(
                false, a, false, bb, (short)0, acc[t], false, false);
        }
    }
    {
        float p[8];
#pragma unroll
        for (int vg = 0; vg < 8; ++vg) p[vg] = 0.f;
#pragma unroll
        for (int t = 0; t < 4; ++t) {
            int c = 64 * wv_n + 16 * t + nl;
            float w2 = px2W[c];
            float bi = px1b[c];
#pragma unroll
            for (int vg = 0; vg < 8; ++vg)
                p[vg] += siluf(acc[t][vg] + bi) * w2;
        }
#pragma unroll
        for (int vg = 0; vg < 8; ++vg) {
#pragma unroll
            for (int off = 1; off < 16; off <<= 1)
                p[vg] += __shfl_xor(p[vg], off, 32);
            if (nl == 0) atomicAdd(&wsum[16 * wv_m + vg + 8 * half], p[vg]);
        }
    }
    __syncthreads();
    if (tid < 32 && (e0 + tid) < E) {
        float w = wsum[tid] + px2b[0];
#pragma unroll
        for (int i = 0; i < 3; ++i)
            atomicAdd(&xupd[(long)sRow[tid] * 3 + i], w * sRel[tid * 3 + i]);
    }
}

// ---------------------------------------------------------------------------
// LayerNorm (+opt residual, +opt relu). One wave per row; D in {256,512}.
// ---------------------------------------------------------------------------
__global__ __launch_bounds__(256) void ln_act(float* __restrict__ y,
                                              const float* __restrict__ resid,
                                              const float* __restrict__ g,
                                              const float* __restrict__ bb,
                                              int M, int D, int act, long bsY) {
    int wave = threadIdx.x >> 5, lane = threadIdx.x & 31;
    int r = blockIdx.x * 8 + wave;
    if (r >= M) return;
    y += (long)blockIdx.z * bsY + (long)r * D;
    if (resid) resid += (long)blockIdx.z * bsY + (long)r * D;
    float v[16];
    int e = D >> 5;
    float s = 0.f;
    for (int i = 0; i < e; ++i) {
        int c = (i << 5) + lane;
        v[i] = y[c] + (resid ? resid[c] : 0.f);
        s += v[i];
    }
#pragma unroll
    for (int o = 16; o > 0; o >>= 1) s += __shfl_xor(s, o, 32);
    float mean = s / (float)D;
    float ss = 0.f;
    for (int i = 0; i < e; ++i) { float d = v[i] - mean; ss += d * d; }
#pragma unroll
    for (int o = 16; o > 0; o >>= 1) ss += __shfl_xor(ss, o, 32);
    float rstd = rsqrtf(ss / (float)D + LN_EPS);
    for (int i = 0; i < e; ++i) {
        int c = (i << 5) + lane;
        float o = (v[i] - mean) * rstd * g[c] + bb[c];
        if (act) o = fmaxf(o, 0.f);
        y[c] = o;
    }
}

// ---------------------------------------------------------------------------
// Small elementwise kernels
// ---------------------------------------------------------------------------
__global__ void x_update(float* __restrict__ x, const float* __restrict__ xupd) {
    int i = blockIdx.x * 256 + threadIdx.x;
    if (i >= LN_) return;
    x    += (long)blockIdx.z * LN_ * 3;
    xupd += (long)blockIdx.z * LN_ * 3;
    int a = i < 20 ? i : 20;
    int b = (LN_ - 1 - i) < 20 ? (LN_ - 1 - i) : 20;
    float di = 1.f / (float)(a + b);
#pragma unroll
    for (int d = 0; d < 3; ++d) x[i * 3 + d] += xupd[i * 3 + d] * di;
}

__global__ void apply_offset(const float* __restrict__ x, const float* __restrict__ pred,
                             float* __restrict__ xo, float base_len) {
    int i = blockIdx.x * 256 + threadIdx.x;
    if (i >= LN_) return;
    x    += (long)blockIdx.z * LN_ * 3;
    pred += (long)blockIdx.z * LN_ * 4;
    xo   += (long)blockIdx.z * LN_ * 3;
    float d0 = pred[i * 4 + 0], d1 = pred[i * 4 + 1], d2 = pred[i * 4 + 2];
    float la = pred[i * 4 + 3];
    float len = base_len + (sigmf(la) - 0.5f) * 0.1f;
    float nrm = fmaxf(sqrtf(d0 * d0 + d1 * d1 + d2 * d2), 1e-12f);
    float inv = len / nrm;
    xo[i * 3 + 0] = x[i * 3 + 0] + d0 * inv;
    xo[i * 3 + 1] = x[i * 3 + 1] + d1 * inv;
    xo[i * 3 + 2] = x[i * 3 + 2] + d2 * inv;
}

__global__ void bond_proj(float* __restrict__ xn, const float* __restrict__ xc) {
    int i = blockIdx.x * 256 + threadIdx.x + 1;
    if (i >= LN_) return;
    xn += (long)blockIdx.z * LN_ * 3;
    xc += (long)blockIdx.z * LN_ * 3;
    float v0 = xn[i * 3 + 0] - xc[(i - 1) * 3 + 0];
    float v1 = xn[i * 3 + 1] - xc[(i - 1) * 3 + 1];
    float v2 = xn[i * 3 + 2] - xc[(i - 1) * 3 + 2];
    float dist = sqrtf(v0 * v0 + v1 * v1 + v2 * v2);
    float lo = 1.33f - 0.02f, hi = 1.33f + 0.02f;
    if (dist < lo || dist > hi) {
        float cl = fminf(fmaxf(dist, lo), hi);
        float s = cl / (dist + 1e-8f);
        xn[i * 3 + 0] = xc[(i - 1) * 3 + 0] + v0 * s;
        xn[i * 3 + 1] = xc[(i - 1) * 3 + 1] + v1 * s;
        xn[i * 3 + 2] = xc[(i - 1) * 3 + 2] + v2 * s;
    }
}

// ---------------------------------------------------------------------------
extern "C" void kernel_launch(void* const* d_in, const int* in_sizes, int n_in,
                              void* d_out, int out_size, void* d_ws, size_t ws_size,
                              hipStream_t stream) {
    const float* zg = (const float*)d_in[0];
    const float* zl = (const float*)d_in[1];
    const int*   ei = (const int*)d_in[2];
    const int E = in_sizes[2] / 2;
    const int* erow = ei;
    const int* ecol = ei + E;
    // params flattened in jax pytree order (dict keys sorted; 'W' < 'b')
    const float* coff1W=(const float*)d_in[3],  *coff1b=(const float*)d_in[4];
    const float* coff2W=(const float*)d_in[5],  *coff2b=(const float*)d_in[6];
    const float* embW  =(const float*)d_in[7],  *embb  =(const float*)d_in[8];
    const float* Lln_b =(const float*)d_in[9],  *Lln_g =(const float*)d_in[10];
    const float* pe1W  =(const float*)d_in[11], *pe1b  =(const float*)d_in[12];
    const float* pe2W  =(const float*)d_in[13], *pe2b  =(const float*)d_in[14];
    const float* ph1W  =(const float*)d_in[15], *ph1b  =(const float*)d_in[16];
    const float* ph2W  =(const float*)d_in[17], *ph2b  =(const float*)d_in[18];
    const float* px1W  =(const float*)d_in[19], *px1b  =(const float*)d_in[20];
    const float* px2W  =(const float*)d_in[21], *px2b  =(const float*)d_in[22];
    const float* ltc1W =(const float*)d_in[23], *ltc1b =(const float*)d_in[24];
    const float* ltc2W =(const float*)d_in[25], *ltc2b =(const float*)d_in[26];
    const float* ltc3W =(const float*)d_in[27], *ltc3b =(const float*)d_in[28];
    const float* ltcln_b=(const float*)d_in[29],*ltcln_g=(const float*)d_in[30];
    const float* noff1W=(const float*)d_in[31], *noff1b=(const float*)d_in[32];
    const float* noff2W=(const float*)d_in[33], *noff2b=(const float*)d_in[34];
    const float* seq1W =(const float*)d_in[35], *seq1b =(const float*)d_in[36];
    const float* seq2W =(const float*)d_in[37], *seq2b =(const float*)d_in[38];
    const float* seq3W =(const float*)d_in[39], *seq3b =(const float*)d_in[40];
    const float* sln1b =(const float*)d_in[41], *sln1g =(const float*)d_in[42];
    const float* sln2b =(const float*)d_in[43], *sln2g =(const float*)d_in[44];

    // ---- workspace carve: f32 pool then f16 pool (16B-aligned throughout)
    float* ws = (float*)d_ws;
    size_t off = 0;
    auto alloc = [&](size_t n) { float* p = ws + off; off += n; return p; };
    float* h    = alloc((size_t)NB * LN_ * 256);
    float* x    = alloc((size_t)NB * LN_ * 3);
    float* t512 = alloc((size_t)NB * LN_ * 512);  // seq s1 / hu
    float* tA   = alloc((size_t)NB * LN_ * 256);  // ltc/seq mid, ph1 out
    float* agg  = alloc((size_t)NB * LN_ * 256);
    float* t128 = alloc((size_t)NB * LN_ * 128);
    float* xupd = alloc((size_t)NB * LN_ * 3);
    float* pred = alloc((size_t)NB * LN_ * 4);

    _Float16* hp = (_Float16*)(ws + off);
    size_t hoff = 0;
    auto halloc = [&](size_t n) { _Float16* p = hp + hoff; hoff += n; return p; };
    _Float16* pe1t  = halloc((size_t)8 * 256 * 544);
    _Float16* pe2t  = halloc((size_t)8 * 256 * 256);
    _Float16* ph1t  = halloc((size_t)8 * 256 * 512);
    _Float16* ph2t  = halloc((size_t)8 * 256 * 256);
    _Float16* px1t  = halloc((size_t)8 * 256 * 256);
    _Float16* embt  = halloc((size_t)256 * 192);
    _Float16* ltc1t = halloc((size_t)256 * 192);
    _Float16* ltc2t = halloc((size_t)128 * 256);
    _Float16* ltc3t = halloc((size_t)64 * 128);   // N=3  padded to 64
    _Float16* seq1t = halloc((size_t)512 * 192);
    _Float16* seq2t = halloc((size_t)256 * 512);
    _Float16* seq3t = halloc((size_t)64 * 256);   // N=20 padded to 64
    _Float16* noff1t= halloc((size_t)128 * 256);
    _Float16* noff2t= halloc((size_t)64 * 128);   // N=4  padded to 64
    _Float16* coff1t= halloc((size_t)128 * 256);
    _Float16* coff2t= halloc((size_t)64 * 128);   // N=4  padded to 64

    auto convT = [&](const float* W, _Float16* Wt, int K, int N, int Kpad, int Npad, int nl) {
        long tot = (long)Npad * Kpad;
        dim3 g((unsigned)((tot + 255) / 256), nl);
        wconvT<<<g, 256, 0, stream>>>(W, Wt, K, N, Kpad, Npad);
    };
    convT(pe1W, pe1t, 513, 256, 544, 256, 8);
    convT(pe2W, pe2t, 256, 256, 256, 256, 8);
    convT(ph1W, ph1t, 512, 256, 512, 256, 8);
    convT(ph2W, ph2t, 256, 256, 256, 256, 8);
    convT(px1W, px1t, 256, 256, 256, 256, 8);
    convT(embW,  embt,  192, 256, 192, 256, 1);
    convT(ltc1W, ltc1t, 192, 256, 192, 256, 1);
    convT(ltc2W, ltc2t, 256, 128, 256, 128, 1);
    convT(ltc3W, ltc3t, 128, 3,   128, 64,  1);
    convT(seq1W, seq1t, 192, 512, 192, 512, 1);
    convT(seq2W, seq2t, 512, 256, 512, 256, 1);
    convT(seq3W, seq3t, 256, 20,  256, 64,  1);
    convT(noff1W, noff1t, 256, 128, 256, 128, 1);
    convT(noff2W, noff2t, 128, 4,   128, 64,  1);
    convT(coff1W, coff1t, 256, 128, 256, 128, 1);
    convT(coff2W, coff2t, 128, 4,   128, 64,  1);

    float* out_xn  = (float*)d_out;                       // B*L*3
    float* out_x   = out_xn + (size_t)NB * LN_ * 3;
    float* out_xc  = out_x  + (size_t)NB * LN_ * 3;
    float* out_seq = out_xc + (size_t)NB * LN_ * 3;       // B*L*20

    const long S256 = (long)LN_ * 256, S512 = (long)LN_ * 512;
    const long S128 = (long)LN_ * 128, S3 = (long)LN_ * 3, S4 = (long)LN_ * 4;
    auto grid = [](int M, int N) { return dim3((M + 127) / 128, (N + 63) / 64, NB); };
    dim3 rowg(LN_ / 8, 1, NB), eltg(LN_ / 256, 1, NB);

    // h = z @ emb  (z = [bcast zg | zl], lda0 = 0 broadcast)
    wgemm<0><<<grid(LN_, 256), 256, 0, stream>>>(zg, 0L, 128, 128L, zl, 64L, (long)LN_ * 64,
                                                 embt, embb, h, LN_, 192, 256, 192, S256);
    // ltc branch -> x
    wgemm<0><<<grid(LN_, 256), 256, 0, stream>>>(zg, 0L, 128, 128L, zl, 64L, (long)LN_ * 64,
                                                 ltc1t, ltc1b, tA, LN_, 192, 256, 192, S256);
    ln_act<<<rowg, 256, 0, stream>>>(tA, nullptr, ltcln_g, ltcln_b, LN_, 256, 1, S256);
    wgemm<1><<<grid(LN_, 128), 256, 0, stream>>>(tA, 256L, 256, S256, tA, 0L, 0L,
                                                 ltc2t, ltc2b, t128, LN_, 256, 128, 256, S128);
    wgemm<0><<<grid(LN_, 3), 256, 0, stream>>>(t128, 128L, 128, S128, t128, 0L, 0L,
                                               ltc3t, ltc3b, x, LN_, 128, 3, 128, S3);
    // seq branch -> out_seq
    wgemm<0><<<grid(LN_, 512), 256, 0, stream>>>(zg, 0L, 128, 128L, zl, 64L, (long)LN_ * 64,
                                                 seq1t, seq1b, t512, LN_, 192, 512, 192, S512);
    ln_act<<<rowg, 256, 0, stream>>>(t512, nullptr, sln1g, sln1b, LN_, 512, 1, S512);
    wgemm<0><<<grid(LN_, 256), 256, 0, stream>>>(t512, 512L, 512, S512, t512, 0L, 0L,
                                                 seq2t, seq2b, tA, LN_, 512, 256, 512, S256);
    ln_act<<<rowg, 256, 0, stream>>>(tA, nullptr, sln2g, sln2b, LN_, 256, 1, S256);
    wgemm<0><<<grid(LN_, 20), 256, 0, stream>>>(tA, 256L, 256, S256, tA, 0L, 0L,
                                                seq3t, seq3b, out_seq, LN_, 256, 20, 256, (long)LN_ * 20);

    // message-passing layers
    dim3 egrid((E + 31) / 32, NB);
    for (int l = 0; l < 8; ++l) {
        hipMemsetAsync(agg, 0, (size_t)NB * LN_ * 256 * sizeof(float), stream);
        hipMemsetAsync(xupd, 0, (size_t)NB * LN_ * 3 * sizeof(float), stream);
        edge_fused<<<egrid, 256, 0, stream>>>(
            h, x, erow, ecol,
            pe1t + (long)l * 256 * 544, pe1b + l * 256,
            pe2t + (long)l * 256 * 256, pe2b + l * 256,
            px1t + (long)l * 256 * 256, px1b + l * 256,
            px2W + (long)l * 256,       px2b + l,
            agg, xupd, E);
        // hu = silu([h|agg] @ ph1) @ ph2 ; h = LN(h + hu)
        wgemm<2><<<grid(LN_, 256), 256, 0, stream>>>(h, 256L, 256, S256, agg, 256L, S256,
                                                     ph1t + (long)l * 256 * 512, ph1b + l * 256,
                                                     tA, LN_, 512, 256, 512, S256);
        wgemm<0><<<grid(LN_, 256), 256, 0, stream>>>(tA, 256L, 256, S256, tA, 0L, 0L,
                                                     ph2t + (long)l * 256 * 256, ph2b + l * 256,
                                                     t512, LN_, 256, 256, 256, S256);
        ln_act<<<rowg, 256, 0, stream>>>(h, t512, Lln_g + l * 256, Lln_b + l * 256, LN_, 256, 0, S256);
        x_update<<<eltg, 256, 0, stream>>>(x, xupd);
    }

    // offsets + outputs
    wgemm<1><<<grid(LN_, 128), 256, 0, stream>>>(h, 256L, 256, S256, h, 0L, 0L,
                                                 noff1t, noff1b, t128, LN_, 256, 128, 256, S128);
    wgemm<0><<<grid(LN_, 4), 256, 0, stream>>>(t128, 128L, 128, S128, t128, 0L, 0L,
                                               noff2t, noff2b, pred, LN_, 128, 4, 128, S4);
    apply_offset<<<eltg, 256, 0, stream>>>(x, pred, out_xn, 1.46f);
    wgemm<1><<<grid(LN_, 128), 256, 0, stream>>>(h, 256L, 256, S256, h, 0L, 0L,
                                                 coff1t, coff1b, t128, LN_, 256, 128, 256, S128);
    wgemm<0><<<grid(LN_, 4), 256, 0, stream>>>(t128, 128L, 128, S128, t128, 0L, 0L,
                                               coff2t, coff2b, pred, LN_, 128, 4, 128, S4);
    apply_offset<<<eltg, 256, 0, stream>>>(x, pred, out_xc, 1.52f);
    hipMemcpyAsync(out_x, x, (size_t)NB * LN_ * 3 * sizeof(float),
                   hipMemcpyDeviceToDevice, stream);
    bond_proj<<<eltg, 256, 0, stream>>>(out_xn, out_xc);
    (void)n_in; (void)out_size; (void)ws_size;
}
